// GlobalStateMem_76630806495886
// MI455X (gfx1250) — matrix operand order repaired and checked by reading, not compiled, not run on hardware
//
#include <hip/hip_runtime.h>
#include <hip/hip_bf16.h>
#include <math.h>

// Problem dims
#define T_DIM 256
#define B_DIM 512
#define E_DIM 256
#define A_DIM 7
#define M_DIM 200
#define S_DIM 30
#define H_DIM 200
#define EA_DIM 263   // E + A
#define G3M 600      // 3*M

// Padded dims (K to mult of 32 for WMMA, N to mult of 16)
#define KP1 288      // 263 -> 288
#define KP2 224      // 200 -> 224
#define NP1 208      // 200 -> 208 (13 tiles)
#define NP2 608      // 600 -> 608 (38 tiles)
#define NP3 64       // 60  -> 64  (4 tiles)

typedef __attribute__((ext_vector_type(16))) _Float16 v16h;
typedef __attribute__((ext_vector_type(8)))  _Float16 v8h;
typedef __attribute__((ext_vector_type(8)))  float    v8f;

// ---------------------------------------------------------------------------
// WMMA fragment loaders (contiguous k order shared by A and B, not the ISA table order; wave32)
// A 16x32 f16: lane l<16 -> M=l, K = k0+{0..15}
//              lane l>=16 -> M=l-16, K = k0+{16..31}
// ---------------------------------------------------------------------------
__device__ __forceinline__ v16h load_a_frag(const _Float16* base, int lda, int k0, int lane) {
    int row = lane & 15;
    int ks  = (lane >> 4) * 16;
    const _Float16* p = base + row * lda + k0 + ks;
    v8h lo = *(const v8h*)(p);
    v8h hi = *(const v8h*)(p + 8);
    v16h a;
#pragma unroll
    for (int i = 0; i < 8; ++i) { a[i] = lo[i]; a[i + 8] = hi[i]; }
    return a;
}

// B 32x16 f16: lane l<16 -> N=l, K = k0+{0..15}; lane l>=16 -> N=l-16, K = k0+{16..31}
// Weight packed as Wp[outRow][Kpad] f16, so B column n == Wp row (n0+n): contiguous.
__device__ __forceinline__ v16h load_b_frag(const _Float16* w, int ldb, int n0, int k0, int lane) {
    int col = lane & 15;
    int kh  = (lane >> 4) * 16;
    return *(const v16h*)(w + (long)(n0 + col) * ldb + k0 + kh);
}

// Fast hardware transcendentals
__device__ __forceinline__ float rcp_f(float x)     { return __builtin_amdgcn_rcpf(x); }
__device__ __forceinline__ float elu_f(float x)     { return x > 0.f ? x : (__expf(x) - 1.f); }
__device__ __forceinline__ float sigmoid_f(float x) { return rcp_f(1.f + __expf(-x)); }
__device__ __forceinline__ float tanh_f(float x) {
    float e = __expf(2.f * x);
    return (e - 1.f) * rcp_f(e + 1.f);
}

// ---------------------------------------------------------------------------
// Pack f32 weight [R][C] -> zero-padded f16 [Rp][Cp]
// ---------------------------------------------------------------------------
__global__ void pack_weight_kernel(_Float16* __restrict__ dst, const float* __restrict__ src,
                                   int R, int C, int Rp, int Cp) {
    int i = blockIdx.x * 256 + threadIdx.x;
    if (i >= Rp * Cp) return;
    int r = i / Cp, c = i % Cp;
    float v = (r < R && c < C) ? src[r * C + c] : 0.f;
    dst[i] = (_Float16)v;
}

// ---------------------------------------------------------------------------
// Phase 1 (parallel over T*B): gx = elu([e,a]@W1^T + b1) @ Wih^T + bih
// One WG = 32 rows (2 row-tiles): each B fragment is loaded once and used by
// two WMMAs, halving weight-fragment L2 traffic vs 16-row blocking.
// ---------------------------------------------------------------------------
__global__ __launch_bounds__(256) void phase1_gx_kernel(
    const float* __restrict__ embed, const float* __restrict__ action,
    const _Float16* __restrict__ W1h, const float* __restrict__ b1,
    const _Float16* __restrict__ Wihh, const float* __restrict__ bih,
    float* __restrict__ GX)
{
    __shared__ __align__(32) _Float16 sXA[32 * KP1];
    __shared__ __align__(32) _Float16 sX[32 * KP2];
    int tid = threadIdx.x, lane = tid & 31, wave = tid >> 5;
    long rowBase = (long)blockIdx.x * 32;

    for (int i = tid; i < 32 * KP1; i += 256) sXA[i] = (_Float16)0.f;
    for (int i = tid; i < 32 * KP2; i += 256) sX[i]  = (_Float16)0.f;
    __syncthreads();

    // stage concat(embed, action) -> f16
    for (int i = tid; i < 32 * EA_DIM; i += 256) {
        int m = i / EA_DIM, k = i % EA_DIM;
        long g = rowBase + m;
        float v = (k < E_DIM) ? embed[g * E_DIM + k] : action[g * A_DIM + (k - E_DIM)];
        sXA[m * KP1 + k] = (_Float16)v;
    }
    __syncthreads();

    // GEMM1: x = elu(XA @ W1^T + b1)  [32 x 200]
    for (int j = wave; j < NP1 / 16; j += 8) {
        v8f acc0 = {}, acc1 = {};
        for (int k0 = 0; k0 < KP1; k0 += 32) {
            v16h b  = load_b_frag(W1h, KP1, j * 16, k0, lane);
            v16h a0 = load_a_frag(sXA,            KP1, k0, lane);
            v16h a1 = load_a_frag(sXA + 16 * KP1, KP1, k0, lane);
            acc0 = __builtin_amdgcn_wmma_f32_16x16x32_f16(false, a0, false, b, (short)0, acc0, false, false);
            acc1 = __builtin_amdgcn_wmma_f32_16x16x32_f16(false, a1, false, b, (short)0, acc1, false, false);
        }
        int n = lane & 15, m0 = (lane >> 4) * 8, col = j * 16 + n;
        if (col < H_DIM) {
            float bb = b1[col];
#pragma unroll
            for (int r = 0; r < 8; ++r) {
                sX[(m0 + r) * KP2 + col]      = (_Float16)elu_f(acc0[r] + bb);
                sX[(16 + m0 + r) * KP2 + col] = (_Float16)elu_f(acc1[r] + bb);
            }
        }
    }
    __syncthreads();

    // GEMM2: gx = x @ Wih^T + bih  [32 x 600]
    for (int j = wave; j < NP2 / 16; j += 8) {
        v8f acc0 = {}, acc1 = {};
        for (int k0 = 0; k0 < KP2; k0 += 32) {
            v16h b  = load_b_frag(Wihh, KP2, j * 16, k0, lane);
            v16h a0 = load_a_frag(sX,            KP2, k0, lane);
            v16h a1 = load_a_frag(sX + 16 * KP2, KP2, k0, lane);
            acc0 = __builtin_amdgcn_wmma_f32_16x16x32_f16(false, a0, false, b, (short)0, acc0, false, false);
            acc1 = __builtin_amdgcn_wmma_f32_16x16x32_f16(false, a1, false, b, (short)0, acc1, false, false);
        }
        int n = lane & 15, m0 = (lane >> 4) * 8, col = j * 16 + n;
        if (col < G3M) {
            float bb = bih[col];
#pragma unroll
            for (int r = 0; r < 8; ++r) {
                GX[(rowBase + m0 + r) * G3M + col]      = acc0[r] + bb;
                GX[(rowBase + 16 + m0 + r) * G3M + col] = acc1[r] + bb;
            }
        }
    }
}

// ---------------------------------------------------------------------------
// Phase 2: single persistent scan. One WG owns 16 batch rows for all T steps.
// 640 threads = 20 waves: each wave owns <=2 column tiles, whose Whh B-frags
// are PRELOADED INTO VGPRS once (2 tiles x 7 ksteps x 8 VGPRs = 112 VGPRs).
// Inside the 256-step loop the WMMA phase touches no global weights at all.
// ---------------------------------------------------------------------------
#define SCAN_THREADS 640
#define SCAN_WAVES   20

__global__ __launch_bounds__(SCAN_THREADS) void gru_scan_kernel(
    const float* __restrict__ in_state, const unsigned char* __restrict__ reset,
    const float* __restrict__ GX, const _Float16* __restrict__ Whhh,
    const float* __restrict__ bhh, float* __restrict__ states)
{
    __shared__ __align__(32) _Float16 sH16[16 * KP2]; // f16 h (K-padded) for WMMA A
    __shared__ float sHf[16 * M_DIM];                 // f32 carried state
    __shared__ float sGH[16 * NP2];                   // gh result
    int tid = threadIdx.x, lane = tid & 31, wave = tid >> 5;
    int rowBase = blockIdx.x * 16;

    // Preload this wave's Whh fragments into registers (wave-uniform guards
    // keep EXEC all-ones for every WMMA, as the ISA requires).
    int j0 = wave, j1 = wave + SCAN_WAVES;
    v16h b0[KP2 / 32], b1[KP2 / 32];
#pragma unroll
    for (int kk = 0; kk < KP2 / 32; ++kk)
        b0[kk] = load_b_frag(Whhh, KP2, j0 * 16, kk * 32, lane);
    if (j1 < NP2 / 16) {
#pragma unroll
        for (int kk = 0; kk < KP2 / 32; ++kk)
            b1[kk] = load_b_frag(Whhh, KP2, j1 * 16, kk * 32, lane);
    }

    for (int i = tid; i < 16 * KP2; i += SCAN_THREADS) sH16[i] = (_Float16)0.f;
    for (int i = tid; i < 16 * M_DIM; i += SCAN_THREADS) {
        int m = i / M_DIM, k = i % M_DIM;
        sHf[i] = in_state[(rowBase + m) * M_DIM + k];
    }
    __syncthreads();

    for (int t = 0; t < T_DIM; ++t) {
        const unsigned char* rs = reset + (size_t)t * B_DIM;
        const float* GXt = GX + (size_t)t * B_DIM * G3M;
        float* hout = states + (size_t)t * B_DIM * M_DIM;

        // h (post-reset) -> f16 A operand
        for (int i = tid; i < 16 * M_DIM; i += SCAN_THREADS) {
            int m = i / M_DIM, k = i % M_DIM;
            float v = rs[rowBase + m] ? 0.f : sHf[m * M_DIM + k];
            sH16[m * KP2 + k] = (_Float16)v;
        }
        __syncthreads();

        // gh = h @ Whh^T  [16 x 600] -- B operands live in VGPRs
        {
            v8f acc0 = {};
#pragma unroll
            for (int kk = 0; kk < KP2 / 32; ++kk) {
                v16h a = load_a_frag(sH16, KP2, kk * 32, lane);
                acc0 = __builtin_amdgcn_wmma_f32_16x16x32_f16(false, a, false, b0[kk], (short)0, acc0, false, false);
            }
            int n = lane & 15, m0 = (lane >> 4) * 8, col = j0 * 16 + n;
#pragma unroll
            for (int r = 0; r < 8; ++r)
                sGH[(m0 + r) * NP2 + col] = acc0[r];
        }
        if (j1 < NP2 / 16) {
            v8f acc1 = {};
#pragma unroll
            for (int kk = 0; kk < KP2 / 32; ++kk) {
                v16h a = load_a_frag(sH16, KP2, kk * 32, lane);
                acc1 = __builtin_amdgcn_wmma_f32_16x16x32_f16(false, a, false, b1[kk], (short)0, acc1, false, false);
            }
            int n = lane & 15, m0 = (lane >> 4) * 8, col = j1 * 16 + n;
#pragma unroll
            for (int r = 0; r < 8; ++r)
                sGH[(m0 + r) * NP2 + col] = acc1[r];
        }
        __syncthreads();

        // gates + state update (each (m,mm) element touched by exactly one thread)
        for (int i = tid; i < 16 * M_DIM; i += SCAN_THREADS) {
            int m = i / M_DIM, mm = i % M_DIM;
            int b = rowBase + m;
            size_t gxr = (size_t)b * G3M;
            float xr = GXt[gxr + mm];
            float xz = GXt[gxr + M_DIM + mm];
            float xn = GXt[gxr + 2 * M_DIM + mm];
            if (t + 1 < T_DIM) {
                // prefetch next timestep's gx slice (streams from HBM; GX >> L2)
                const float* nx = GXt + (size_t)B_DIM * G3M + gxr;
                __builtin_prefetch(nx + mm, 0, 0);
                __builtin_prefetch(nx + M_DIM + mm, 0, 0);
                __builtin_prefetch(nx + 2 * M_DIM + mm, 0, 0);
            }
            float hr = sGH[m * NP2 + mm]             + bhh[mm];
            float hz = sGH[m * NP2 + M_DIM + mm]     + bhh[M_DIM + mm];
            float hn = sGH[m * NP2 + 2 * M_DIM + mm] + bhh[2 * M_DIM + mm];
            float rg = sigmoid_f(xr + hr);
            float z  = sigmoid_f(xz + hz);
            float nn = tanh_f(xn + rg * hn);
            float hv = rs[b] ? 0.f : sHf[m * M_DIM + mm];
            float hnew = (1.f - z) * nn + z * hv;
            hout[b * M_DIM + mm] = hnew;
            sHf[m * M_DIM + mm]  = hnew;
        }
        __syncthreads();
    }
}

// ---------------------------------------------------------------------------
// Phase 3 (parallel over T*B): posts = elu(states@W2^T + b2) @ W3^T + b3
// 32-row blocking, B-frag reused across two row tiles.
// ---------------------------------------------------------------------------
__global__ __launch_bounds__(256) void post_mlp_kernel(
    const float* __restrict__ states,
    const _Float16* __restrict__ W2h, const float* __restrict__ b2,
    const _Float16* __restrict__ W3h, const float* __restrict__ b3,
    float* __restrict__ posts)
{
    __shared__ __align__(32) _Float16 sS[32 * KP2];
    __shared__ __align__(32) _Float16 sP[32 * KP2];
    int tid = threadIdx.x, lane = tid & 31, wave = tid >> 5;
    long rowBase = (long)blockIdx.x * 32;

    for (int i = tid; i < 32 * KP2; i += 256) { sS[i] = (_Float16)0.f; sP[i] = (_Float16)0.f; }
    __syncthreads();
    for (int i = tid; i < 32 * M_DIM; i += 256) {
        int m = i / M_DIM, k = i % M_DIM;
        sS[m * KP2 + k] = (_Float16)states[(rowBase + m) * M_DIM + k];
    }
    __syncthreads();

    // p = elu(h @ W2^T + b2)  [32 x 200]
    for (int j = wave; j < NP1 / 16; j += 8) {
        v8f acc0 = {}, acc1 = {};
        for (int k0 = 0; k0 < KP2; k0 += 32) {
            v16h b  = load_b_frag(W2h, KP2, j * 16, k0, lane);
            v16h a0 = load_a_frag(sS,            KP2, k0, lane);
            v16h a1 = load_a_frag(sS + 16 * KP2, KP2, k0, lane);
            acc0 = __builtin_amdgcn_wmma_f32_16x16x32_f16(false, a0, false, b, (short)0, acc0, false, false);
            acc1 = __builtin_amdgcn_wmma_f32_16x16x32_f16(false, a1, false, b, (short)0, acc1, false, false);
        }
        int n = lane & 15, m0 = (lane >> 4) * 8, col = j * 16 + n;
        if (col < H_DIM) {
            float bb = b2[col];
#pragma unroll
            for (int r = 0; r < 8; ++r) {
                sP[(m0 + r) * KP2 + col]      = (_Float16)elu_f(acc0[r] + bb);
                sP[(16 + m0 + r) * KP2 + col] = (_Float16)elu_f(acc1[r] + bb);
            }
        }
    }
    __syncthreads();

    // post = p @ W3^T + b3  [32 x 60]
    for (int j = wave; j < NP3 / 16; j += 8) {
        v8f acc0 = {}, acc1 = {};
        for (int k0 = 0; k0 < KP2; k0 += 32) {
            v16h b  = load_b_frag(W3h, KP2, j * 16, k0, lane);
            v16h a0 = load_a_frag(sP,            KP2, k0, lane);
            v16h a1 = load_a_frag(sP + 16 * KP2, KP2, k0, lane);
            acc0 = __builtin_amdgcn_wmma_f32_16x16x32_f16(false, a0, false, b, (short)0, acc0, false, false);
            acc1 = __builtin_amdgcn_wmma_f32_16x16x32_f16(false, a1, false, b, (short)0, acc1, false, false);
        }
        int n = lane & 15, m0 = (lane >> 4) * 8, col = j * 16 + n;
        if (col < 2 * S_DIM) {
            float bb = b3[col];
#pragma unroll
            for (int r = 0; r < 8; ++r) {
                posts[(rowBase + m0 + r) * (2 * S_DIM) + col]      = acc0[r] + bb;
                posts[(rowBase + 16 + m0 + r) * (2 * S_DIM) + col] = acc1[r] + bb;
            }
        }
    }
}

// ---------------------------------------------------------------------------
// Final sample: mean + (softplus(std_raw)+0.1) * N(0,1)
// ---------------------------------------------------------------------------
__device__ __forceinline__ unsigned hash_u32(unsigned x) {
    x ^= x >> 17; x *= 0xed5ad4bbu; x ^= x >> 11; x *= 0xac4c1b51u;
    x ^= x >> 15; x *= 0x31848babu; x ^= x >> 14; return x;
}

__global__ void sample_kernel(const float* __restrict__ posts, float* __restrict__ sample) {
    int i = blockIdx.x * blockDim.x + threadIdx.x;
    if (i >= B_DIM * S_DIM) return;
    int b = i / S_DIM, s = i % S_DIM;
    long base = ((long)(T_DIM - 1) * B_DIM + b) * (2 * S_DIM);
    float mean = posts[base + s];
    float sr   = posts[base + S_DIM + s];
    float sp   = sr > 20.f ? sr : log1pf(__expf(sr));
    float stdv = sp + 0.1f;
    unsigned h1 = hash_u32(0x2a2a2a2au ^ (unsigned)i);
    unsigned h2 = hash_u32(h1 + 0x85ebca6bu);
    float u1 = ((h1 >> 8) + 1u) * (1.f / 16777217.f);
    float u2 = (h2 >> 8) * (1.f / 16777216.f);
    float z  = sqrtf(-2.f * __logf(u1)) * __cosf(6.28318530718f * u2);
    sample[i] = mean + stdv * z;
}

// ---------------------------------------------------------------------------
// Workspace layout (all offsets multiples of 256B -> 32B alignment for v16h)
// ---------------------------------------------------------------------------
#define WS_W1H   0UL
#define WS_WIHH  (WS_W1H  + (size_t)NP1 * KP1 * 2)
#define WS_WHHH  (WS_WIHH + (size_t)NP2 * KP2 * 2)
#define WS_W2H   (WS_WHHH + (size_t)NP2 * KP2 * 2)
#define WS_W3H   (WS_W2H  + (size_t)NP1 * KP2 * 2)
#define WS_GX    (WS_W3H  + (size_t)NP3 * KP2 * 2)
// GX: T*B*600 f32 = 314.6 MB

extern "C" void kernel_launch(void* const* d_in, const int* in_sizes, int n_in,
                              void* d_out, int out_size, void* d_ws, size_t ws_size,
                              hipStream_t stream) {
    const float* embed    = (const float*)d_in[0];
    const float* action   = (const float*)d_in[1];
    const unsigned char* reset = (const unsigned char*)d_in[2];
    const float* in_state = (const float*)d_in[3];
    const float* W1  = (const float*)d_in[4];
    const float* b1  = (const float*)d_in[5];
    const float* Wih = (const float*)d_in[6];
    const float* bih = (const float*)d_in[7];
    const float* Whh = (const float*)d_in[8];
    const float* bhh = (const float*)d_in[9];
    const float* W2  = (const float*)d_in[10];
    const float* b2  = (const float*)d_in[11];
    const float* W3  = (const float*)d_in[12];
    const float* b3  = (const float*)d_in[13];

    float* out    = (float*)d_out;
    float* sample = out;                                        // [B, S]
    float* states = out + (size_t)B_DIM * S_DIM;                // [T, B, M]
    float* posts  = states + (size_t)T_DIM * B_DIM * M_DIM;     // [T, B, 2S]

    char* ws = (char*)d_ws;
    _Float16* W1h  = (_Float16*)(ws + WS_W1H);
    _Float16* Wihh = (_Float16*)(ws + WS_WIHH);
    _Float16* Whhh = (_Float16*)(ws + WS_WHHH);
    _Float16* W2h  = (_Float16*)(ws + WS_W2H);
    _Float16* W3h  = (_Float16*)(ws + WS_W3H);
    float*    GX   = (float*)   (ws + WS_GX);

    // Pack weights to padded f16
    auto packGrid = [](int n) { return dim3((n + 255) / 256); };
    pack_weight_kernel<<<packGrid(NP1 * KP1), 256, 0, stream>>>(W1h,  W1,  H_DIM, EA_DIM, NP1, KP1);
    pack_weight_kernel<<<packGrid(NP2 * KP2), 256, 0, stream>>>(Wihh, Wih, G3M,   M_DIM,  NP2, KP2);
    pack_weight_kernel<<<packGrid(NP2 * KP2), 256, 0, stream>>>(Whhh, Whh, G3M,   M_DIM,  NP2, KP2);
    pack_weight_kernel<<<packGrid(NP1 * KP2), 256, 0, stream>>>(W2h,  W2,  H_DIM, M_DIM,  NP1, KP2);
    pack_weight_kernel<<<packGrid(NP3 * KP2), 256, 0, stream>>>(W3h,  W3,  2*S_DIM, M_DIM, NP3, KP2);

    // Phase 1: all-timestep input path (parallel over T*B rows, 32 rows/WG)
    phase1_gx_kernel<<<(T_DIM * B_DIM) / 32, 256, 0, stream>>>(
        embed, action, W1h, b1, Wihh, bih, GX);

    // Phase 2: entire sequential scan in ONE launch (batch rows independent)
    gru_scan_kernel<<<B_DIM / 16, SCAN_THREADS, 0, stream>>>(
        in_state, reset, GX, Whhh, bhh, states);

    // Phase 3: post MLP over all T*B rows (parallel, 32 rows/WG)
    post_mlp_kernel<<<(T_DIM * B_DIM) / 32, 256, 0, stream>>>(
        states, W2h, b2, W3h, b3, posts);

    // Final sample
    sample_kernel<<<(B_DIM * S_DIM + 255) / 256, 256, 0, stream>>>(posts, sample);
}
